// SqueezeAttention_13640816132789
// MI455X (gfx1250) — matrix operand order repaired and checked by reading, not compiled, not run on hardware
//
#include <hip/hip_runtime.h>

// ---------------------------------------------------------------------------
// SqueezeAttention on MI455X (gfx1250, wave32, WMMA f32_16x16x32_f16)
// Double-buffered LDS + register staging to hide HBM latency behind WMMA;
// permlane16 cross-lane reductions (no DS round-trips) for online softmax.
// ---------------------------------------------------------------------------
typedef __attribute__((ext_vector_type(16))) _Float16 v16h;
typedef __attribute__((ext_vector_type(8)))  _Float16 v8h;
typedef __attribute__((ext_vector_type(8)))  float    v8f;

#define WMMA_F32_F16(A_, B_, C_) \
  __builtin_amdgcn_wmma_f32_16x16x32_f16(false, (A_), false, (B_), (short)0, (C_), false, false)

constexpr int B_   = 4;
constexpr int T_   = 2048;
constexpr int E_   = 1024;
constexpr int H_   = 16;
constexpr int S_   = 512;   // squeeze dim
constexpr int SHD_ = 32;    // squeezed head dim (== WMMA K, perfect fit)
constexpr int HD_  = 64;    // value head dim

// ---------------------------------------------------------------------------
// xor-shuffle within 16-lane rows. permlane16 = pure VALU (no LDS round-trip);
// fallback to ds_bpermute-based __shfl_xor if builtin unavailable.
// ---------------------------------------------------------------------------
template <int STEP>  // STEP 0..3 -> xor mask 1,2,4,8
__device__ __forceinline__ float lane16_xor(float v) {
#if __has_builtin(__builtin_amdgcn_permlane16)
  constexpr unsigned s0[4] = {0x67452301u, 0x54761032u, 0x32107654u, 0xFEDCBA98u};
  constexpr unsigned s1[4] = {0xEFCDAB89u, 0xDCFE98BAu, 0xBA98FEDCu, 0x76543210u};
  return __int_as_float(__builtin_amdgcn_permlane16(
      __float_as_int(v), __float_as_int(v), s0[STEP], s1[STEP], false, false));
#else
  return __shfl_xor(v, 1 << STEP);
#endif
}

template <int STEP, bool IS_MAX>
__device__ __forceinline__ void red_step(float (&a)[8]) {
  float t[8];
#pragma unroll
  for (int j = 0; j < 8; ++j) t[j] = lane16_xor<STEP>(a[j]);
#pragma unroll
  for (int j = 0; j < 8; ++j) a[j] = IS_MAX ? fmaxf(a[j], t[j]) : (a[j] + t[j]);
}
template <bool IS_MAX>
__device__ __forceinline__ void red16(float (&a)[8]) {  // reduce across 16 lanes
  red_step<0, IS_MAX>(a);
  red_step<1, IS_MAX>(a);
  red_step<2, IS_MAX>(a);
  red_step<3, IS_MAX>(a);
}

// ---------------------------------------------------------------------------
// fp32 -> f16 conversion with scale (folds softmax scaling into Wq)
// ---------------------------------------------------------------------------
__global__ void k_cvt(const float* __restrict__ in, _Float16* __restrict__ out,
                      int n, float scale) {
  int i = blockIdx.x * blockDim.x + threadIdx.x;
  if (i < n) out[i] = (_Float16)(in[i] * scale);
}

// ---------------------------------------------------------------------------
// Tiled GEMM: C[M,N] = A[M,K](f16) * Bw[K,N](f16) + bias*bscale
// 256 thr (8 waves), 128x128 tile, wave w -> rows 16w..16w+15.
// Double-buffered LDS; next chunk staged through VGPRs during compute.
// ---------------------------------------------------------------------------
template <typename OutT>
__global__ __launch_bounds__(256) void k_gemm(
    const _Float16* __restrict__ A, const _Float16* __restrict__ Bw,
    const float* __restrict__ bias, float bscale,
    OutT* __restrict__ C, int M, int N, int K) {
  constexpr int BT_LD = 40;  // padded stride (halfs) -> conflict-free B frags
  __shared__ __align__(16) _Float16 As[2][128 * 32];
  __shared__ __align__(16) _Float16 Bts[2][128 * BT_LD];

  const int tid  = threadIdx.x;
  const int wave = tid >> 5;
  const int lane = tid & 31;
  const int hx   = lane >> 4;
  const int l16  = lane & 15;
  const int m0 = blockIdx.y * 128;
  const int n0 = blockIdx.x * 128;

  const int a_row = tid >> 1,  a_d0 = (tid & 1) * 16;   // A: 128x32 tile
  const int b_k   = tid & 31,  b_c0 = (tid >> 5) * 16;  // B: 32x128 tile

  v8f acc[8] = {};
  v8h ra0, ra1, rb0, rb1;  // register staging for next chunk

  auto issue = [&](int k0) {  // global -> regs
    const _Float16* sa = A + (size_t)(m0 + a_row) * K + k0 + a_d0;
    ra0 = *(const v8h*)(sa);
    ra1 = *(const v8h*)(sa + 8);
    const _Float16* sb = Bw + (size_t)(k0 + b_k) * N + n0 + b_c0;
    rb0 = *(const v8h*)(sb);
    rb1 = *(const v8h*)(sb + 8);
  };
  auto stage = [&](int buf) {  // regs -> LDS (B transposed)
    *(v8h*)&As[buf][a_row * 32 + a_d0]     = ra0;
    *(v8h*)&As[buf][a_row * 32 + a_d0 + 8] = ra1;
#pragma unroll
    for (int i = 0; i < 8; ++i) {
      Bts[buf][(b_c0 + i) * BT_LD + b_k]     = rb0[i];
      Bts[buf][(b_c0 + 8 + i) * BT_LD + b_k] = rb1[i];
    }
  };

  const int niter = K / 32;
  issue(0);
  stage(0);
  __syncthreads();

  for (int i = 0; i < niter; ++i) {
    if (i + 1 < niter) issue((i + 1) * 32);
    if (i + 2 < niter) {  // CDNA5 global_prefetch_b8, two chunks ahead
      __builtin_prefetch(A + (size_t)(m0 + a_row) * K + (i + 2) * 32 + a_d0, 0, 1);
      __builtin_prefetch(Bw + (size_t)((i + 2) * 32 + b_k) * N + n0 + b_c0, 0, 1);
    }

    const _Float16* AsB  = As[i & 1];
    const _Float16* BtsB = Bts[i & 1];

    // A fragment: 16x32 layout — row l16; K {0..15} / {16..31}
    const _Float16* ar = &AsB[(wave * 16 + l16) * 32];
    v8h alo = *(const v8h*)(ar + 16 * hx);
    v8h ahi = *(const v8h*)(ar + 16 * hx + 8);
    v16h afrag;
#pragma unroll
    for (int q = 0; q < 8; ++q) { afrag[q] = alo[q]; afrag[8 + q] = ahi[q]; }

#pragma unroll
    for (int t = 0; t < 8; ++t) {
      const _Float16* br = &BtsB[(t * 16 + l16) * BT_LD + 16 * hx];
      v8h blo = *(const v8h*)(br);
      v8h bhi = *(const v8h*)(br + 8);
      v16h bfrag;
#pragma unroll
      for (int q = 0; q < 8; ++q) { bfrag[q] = blo[q]; bfrag[8 + q] = bhi[q]; }
      acc[t] = WMMA_F32_F16(afrag, bfrag, acc[t]);
    }

    if (i + 1 < niter) stage((i + 1) & 1);
    __syncthreads();
  }

  // epilogue: C layout row = vgpr + 8*hx, col = lane&15
#pragma unroll
  for (int t = 0; t < 8; ++t) {
    const int col = n0 + t * 16 + l16;
    const float bv = bias ? bias[col] * bscale : 0.f;
#pragma unroll
    for (int j = 0; j < 8; ++j) {
      const int row = m0 + wave * 16 + j + 8 * hx;
      C[(size_t)row * N + col] = (OutT)(acc[t][j] + bv);
    }
  }
}

// ---------------------------------------------------------------------------
// Flash attention: one (b,h), 128 query rows per block (8 waves x 16 rows).
// SHD=32 => S = Q*K^T is one 16x16x32 WMMA per tile. Online softmax with
// permlane16 row reductions; P C-layout -> A-layout via per-wave LDS; PV WMMA.
// K/V tiles double-buffered with register staging.
// ---------------------------------------------------------------------------
__global__ __launch_bounds__(256) void k_flash(
    const _Float16* __restrict__ Qh, const _Float16* __restrict__ Kh,
    const _Float16* __restrict__ Vh, _Float16* __restrict__ ctx) {
  constexpr int VT_LD = 72;  // padded strides (multiple of 8 halfs)
  constexpr int PS_LD = 72;
  __shared__ __align__(16) _Float16 Qs[128 * 32];
  __shared__ __align__(16) _Float16 Ks[2][64 * 32];
  __shared__ __align__(16) _Float16 Vts[2][64 * VT_LD];   // [col][key]
  __shared__ __align__(16) _Float16 Ps[8][16 * PS_LD];    // per-wave P scratch

  const int tid  = threadIdx.x;
  const int wave = tid >> 5;
  const int lane = tid & 31;
  const int hx   = lane >> 4;
  const int l16  = lane & 15;
  const int b = blockIdx.x / H_;
  const int h = blockIdx.x % H_;
  const int q0 = blockIdx.y * 128;

  {  // stage Q block 128x32 (resident for whole kernel)
    const int row = tid >> 1, d0 = (tid & 1) * 16;
    const _Float16* src = Qh + (size_t)(b * T_ + q0 + row) * S_ + h * SHD_ + d0;
    *(v8h*)&Qs[row * 32 + d0]     = *(const v8h*)(src);
    *(v8h*)&Qs[row * 32 + d0 + 8] = *(const v8h*)(src + 8);
  }

  const int kv_key = tid >> 2;
  const int k_d0   = (tid & 3) * 8;
  const int v_c0   = (tid & 3) * 16;
  const _Float16* Kbase = Kh + (size_t)(b * T_) * S_ + h * SHD_;
  const _Float16* Vbase = Vh + (size_t)(b * T_) * E_ + h * HD_;

  v8h rk, rv0, rv1;  // register staging
  auto issue = [&](int kb) {
    rk = *(const v8h*)(Kbase + (size_t)(kb + kv_key) * S_ + k_d0);
    const _Float16* vs = Vbase + (size_t)(kb + kv_key) * E_ + v_c0;
    rv0 = *(const v8h*)(vs);
    rv1 = *(const v8h*)(vs + 8);
  };
  auto stage = [&](int buf) {
    *(v8h*)&Ks[buf][kv_key * 32 + k_d0] = rk;
#pragma unroll
    for (int i = 0; i < 8; ++i) {
      Vts[buf][(v_c0 + i) * VT_LD + kv_key]     = rv0[i];
      Vts[buf][(v_c0 + 8 + i) * VT_LD + kv_key] = rv1[i];
    }
  };

  v8f o[4] = {};
  float m_run[8], l_run[8];
#pragma unroll
  for (int j = 0; j < 8; ++j) { m_run[j] = -1e30f; l_run[j] = 0.f; }

  issue(0);
  stage(0);
  __syncthreads();

  // Q fragment (invariant: full contraction K = SHD = 32)
  const _Float16* qr = &Qs[(wave * 16 + l16) * 32];
  v8h qlo = *(const v8h*)(qr + 16 * hx);
  v8h qhi = *(const v8h*)(qr + 16 * hx + 8);
  v16h qa;
#pragma unroll
  for (int q = 0; q < 8; ++q) { qa[q] = qlo[q]; qa[8 + q] = qhi[q]; }

  constexpr int NBLK = T_ / 64;
  for (int it = 0; it < NBLK; ++it) {
    if (it + 1 < NBLK) issue((it + 1) * 64);

    const _Float16* KsB = Ks[it & 1];
    const _Float16* VtB = Vts[it & 1];

    // scores: 4 tiles of 16 keys, one WMMA each
    v8f s[4];
#pragma unroll
    for (int t = 0; t < 4; ++t) {
      const _Float16* kr = &KsB[(t * 16 + l16) * 32 + 16 * hx];
      v8h klo = *(const v8h*)(kr);
      v8h khi = *(const v8h*)(kr + 8);
      v16h kf;
#pragma unroll
      for (int q = 0; q < 8; ++q) { kf[q] = klo[q]; kf[8 + q] = khi[q]; }
      v8f z = {};
      s[t] = WMMA_F32_F16(qa, kf, z);
    }

    // ---- online softmax (row = vgpr j + 8*hx, cols across 16 lanes)
    float mx[8];
#pragma unroll
    for (int j = 0; j < 8; ++j)
      mx[j] = fmaxf(fmaxf(s[0][j], s[1][j]), fmaxf(s[2][j], s[3][j]));
    red16<true>(mx);

    float alpha[8];
#pragma unroll
    for (int j = 0; j < 8; ++j) {
      float mnew = fmaxf(m_run[j], mx[j]);
      alpha[j] = __expf(m_run[j] - mnew);
      m_run[j] = mnew;
    }
#pragma unroll
    for (int t = 0; t < 4; ++t)
#pragma unroll
      for (int j = 0; j < 8; ++j) s[t][j] = __expf(s[t][j] - m_run[j]);

    float rsum[8];
#pragma unroll
    for (int j = 0; j < 8; ++j)
      rsum[j] = (s[0][j] + s[1][j]) + (s[2][j] + s[3][j]);
    red16<false>(rsum);

#pragma unroll
    for (int j = 0; j < 8; ++j) l_run[j] = l_run[j] * alpha[j] + rsum[j];
#pragma unroll
    for (int t = 0; t < 4; ++t)
#pragma unroll
      for (int j = 0; j < 8; ++j) o[t][j] *= alpha[j];

    // ---- P: C-layout -> A-layout via per-wave LDS scratch
#pragma unroll
    for (int t = 0; t < 4; ++t)
#pragma unroll
      for (int j = 0; j < 8; ++j)
        Ps[wave][(j + 8 * hx) * PS_LD + t * 16 + l16] = (_Float16)s[t][j];

    // ---- O += P(16x64) @ V(64x64): 2 k-chunks x 4 output tiles
#pragma unroll
    for (int kk = 0; kk < 64; kk += 32) {
      const _Float16* pr = &Ps[wave][l16 * PS_LD + kk];
      v8h plo = *(const v8h*)(pr + 16 * hx);
      v8h phi = *(const v8h*)(pr + 16 * hx + 8);
      v16h pa;
#pragma unroll
      for (int q = 0; q < 8; ++q) { pa[q] = plo[q]; pa[8 + q] = phi[q]; }
#pragma unroll
      for (int t = 0; t < 4; ++t) {
        const _Float16* vr = &VtB[(t * 16 + l16) * VT_LD + kk + 16 * hx];
        v8h vlo = *(const v8h*)(vr);
        v8h vhi = *(const v8h*)(vr + 8);
        v16h vf;
#pragma unroll
        for (int q = 0; q < 8; ++q) { vf[q] = vlo[q]; vf[8 + q] = vhi[q]; }
        o[t] = WMMA_F32_F16(pa, vf, o[t]);
      }
    }

    if (it + 1 < NBLK) stage((it + 1) & 1);
    __syncthreads();
  }

  // epilogue: normalize, emit context (f16) in [B,T,E] layout
#pragma unroll
  for (int t = 0; t < 4; ++t)
#pragma unroll
    for (int j = 0; j < 8; ++j) {
      const int q   = q0 + wave * 16 + j + 8 * hx;
      const int col = h * HD_ + t * 16 + l16;
      ctx[(size_t)(b * T_ + q) * E_ + col] = (_Float16)(o[t][j] / l_run[j]);
    }
}

// ---------------------------------------------------------------------------
// Host-side launch sequence (stream-ordered, graph-capture safe)
// ---------------------------------------------------------------------------
extern "C" void kernel_launch(void* const* d_in, const int* in_sizes, int n_in,
                              void* d_out, int out_size, void* d_ws, size_t ws_size,
                              hipStream_t stream) {
  (void)in_sizes; (void)n_in; (void)out_size; (void)ws_size;
  const float* hs = (const float*)d_in[0];
  const float* Wq = (const float*)d_in[1];
  const float* bq = (const float*)d_in[2];
  const float* Wk = (const float*)d_in[3];
  const float* bk = (const float*)d_in[4];
  const float* Wv = (const float*)d_in[5];
  const float* bv = (const float*)d_in[6];
  const float* Wo = (const float*)d_in[7];
  const float* bo = (const float*)d_in[8];
  float* out = (float*)d_out;

  char* p = (char*)d_ws;
  auto alloc = [&](size_t halfs) -> _Float16* {
    _Float16* r = (_Float16*)p;
    p += ((halfs * sizeof(_Float16)) + 255) & ~(size_t)255;
    return r;
  };
  _Float16* hs_h = alloc((size_t)B_ * T_ * E_);   // 16 MB
  _Float16* Wq_h = alloc((size_t)E_ * S_);        //  1 MB
  _Float16* Wk_h = alloc((size_t)E_ * S_);        //  1 MB
  _Float16* Wv_h = alloc((size_t)E_ * E_);        //  2 MB
  _Float16* Wo_h = alloc((size_t)E_ * E_);        //  2 MB
  _Float16* Q_h  = alloc((size_t)B_ * T_ * S_);   //  8 MB
  _Float16* K_h  = alloc((size_t)B_ * T_ * S_);   //  8 MB
  _Float16* V_h  = alloc((size_t)B_ * T_ * E_);   // 16 MB
  _Float16* ctx_h = hs_h;  // hs_h dead after V projection -> reuse (~54 MB)

  const float SCALING = 0.17677669529663687f;  // 32^-0.5, folded into Wq/bq

  auto cvt = [&](const float* in, _Float16* o2, size_t n, float sc) {
    k_cvt<<<dim3((unsigned)((n + 255) / 256)), dim3(256), 0, stream>>>(in, o2, (int)n, sc);
  };
  cvt(hs, hs_h, (size_t)B_ * T_ * E_, 1.f);
  cvt(Wq, Wq_h, (size_t)E_ * S_, SCALING);
  cvt(Wk, Wk_h, (size_t)E_ * S_, 1.f);
  cvt(Wv, Wv_h, (size_t)E_ * E_, 1.f);
  cvt(Wo, Wo_h, (size_t)E_ * E_, 1.f);

  const int M = B_ * T_;  // 8192
  // projections (f16 out)
  k_gemm<_Float16><<<dim3(S_ / 128, M / 128), 256, 0, stream>>>(hs_h, Wq_h, bq, SCALING, Q_h, M, S_, E_);
  k_gemm<_Float16><<<dim3(S_ / 128, M / 128), 256, 0, stream>>>(hs_h, Wk_h, bk, 1.f, K_h, M, S_, E_);
  k_gemm<_Float16><<<dim3(E_ / 128, M / 128), 256, 0, stream>>>(hs_h, Wv_h, bv, 1.f, V_h, M, E_, E_);
  // flash attention -> context (f16)
  k_flash<<<dim3(B_ * H_, T_ / 128), 256, 0, stream>>>(Q_h, K_h, V_h, ctx_h);
  // output projection (f32 out, with bias)
  k_gemm<float><<<dim3(E_ / 128, M / 128), 256, 0, stream>>>(ctx_h, Wo_h, bo, 1.f, out, M, E_, E_);
}
